// AutoFlow_84430467105541
// MI455X (gfx1250) — matrix operand order repaired and checked by reading, not compiled, not run on hardware
//
#include <hip/hip_runtime.h>
#include <math.h>

// Masked autoregressive flow, 16 layers, fused per-wave.
// Each single-wave workgroup owns 32 batch rows and runs all 16 layers
// locally: activations in wave-private LDS, weights pre-masked, f16, and
// pre-swizzled into WMMA B-fragment order so every fragment load is one
// coalesced 1KB burst. Two 16-row A sub-tiles share each B fragment
// (2 independent WMMA chains -> hazard slots filled, B traffic halved).

typedef _Float16 v16h __attribute__((ext_vector_type(16)));
typedef _Float16 v8h  __attribute__((ext_vector_type(8)));
typedef float    v8f  __attribute__((ext_vector_type(8)));

#define D_IN   64
#define NHID   256
#define NLAYER 16
#define MROWS  32        // batch rows per wave

#define PW0   (256 * 64)
#define PW1   (256 * 256)
#define PW2   (64 * 256)
#define PNET  (PW0 + PW1 + PW2)   // 98304 halves per net
#define PLAY  (2 * PNET)          // per layer (loc + scale)

// Per-wave LDS layout (bytes)
#define OFF_YF   0        // float [32][64]   : 8192
#define OFF_YH   8192     // f16   [32][64]   : 4096
#define OFF_H0   12288    // f16   [32][256]  : 16384
#define OFF_H1   28672    // f16   [32][256]  : 16384
#define OFF_LOC  45056    // float [32][64]   : 8192
#define WAVE_LDS 53248

union V16U { v16h v; v8h h[2]; };

__device__ __forceinline__ v8f wmma16(const v16h a, const v16h b, v8f acc) {
    return __builtin_amdgcn_wmma_f32_16x16x32_f16(
        false, a, false, b, (short)0, acc, false, false);
}

// Load the A fragment for 16 rows starting at row r0, k-block ks*32,
// from LDS tile A[.][K] (row-major f16). Contiguous k = 16*(lane>=16) + s.
__device__ __forceinline__ v16h a_frag(const _Float16* A, int K, int r0,
                                       int ks, int lane) {
    const int m     = r0 + (lane & 15);
    const int khalf = (lane >> 4) * 16;
    const _Float16* p = A + (size_t)m * K + ks * 32 + khalf;
    V16U a;
    a.h[0] = *(const v8h*)(p);
    a.h[1] = *(const v8h*)(p + 8);
    return a.v;
}

// Hidden stage: H[32][N] = relu(A[32][K] @ W^T + bias), f16 out to LDS.
// Wsw is swizzled: [ntile][kstep][lane][16 halves].
template <int K, int N>
__device__ __forceinline__ void hidden_stage(const _Float16* __restrict__ Wsw,
                                             const float* __restrict__ bias,
                                             const _Float16* A, _Float16* H,
                                             int lane) {
#pragma unroll 1
    for (int nt = 0; nt < N / 16; ++nt) {
        const int col = (lane & 15) + nt * 16;
        const float bb = bias[col];
        v8f acc0 = {bb, bb, bb, bb, bb, bb, bb, bb};
        v8f acc1 = acc0;
        const _Float16* Wb = Wsw + ((size_t)nt * (K / 32)) * 512 + lane * 16;
#pragma unroll
        for (int ks = 0; ks < K / 32; ++ks) {
            const v16h b  = *(const v16h*)(Wb + (size_t)ks * 512);
            const v16h a0 = a_frag(A, K, 0,  ks, lane);
            const v16h a1 = a_frag(A, K, 16, ks, lane);
            acc0 = wmma16(a0, b, acc0);
            acc1 = wmma16(a1, b, acc1);
        }
        const int rbase = (lane >> 4) * 8;
#pragma unroll
        for (int v = 0; v < 8; ++v) {
            const float x0 = fmaxf(acc0[v], 0.0f);   // single v_max_num_f32
            const float x1 = fmaxf(acc1[v], 0.0f);
            H[(size_t)(rbase + v) * N + col]        = (_Float16)x0;
            H[(size_t)(rbase + v + 16) * N + col]   = (_Float16)x1;
        }
    }
}

__global__ void __launch_bounds__(32)
flow_kernel(const float* __restrict__ u,
            const float* __restrict__ lb0, const float* __restrict__ lb1,
            const float* __restrict__ lb2,
            const float* __restrict__ sb0, const float* __restrict__ sb1,
            const float* __restrict__ sb2,
            const _Float16* __restrict__ Wpack,
            float* __restrict__ out) {
    extern __shared__ char smem[];
    const int lane = threadIdx.x & 31;
    const int row0 = blockIdx.x * MROWS;

    float*    yf   = (float*)(smem + OFF_YF);
    _Float16* yh   = (_Float16*)(smem + OFF_YH);
    _Float16* h0   = (_Float16*)(smem + OFF_H0);
    _Float16* h1   = (_Float16*)(smem + OFF_H1);
    float*    locb = (float*)(smem + OFF_LOC);

    // Load this wave's 32x64 input tile (coalesced), f32 master + f16 copy.
#pragma unroll
    for (int i = 0; i < MROWS * D_IN / 32; ++i) {
        const int idx = i * 32 + lane;
        const float v = u[(size_t)row0 * D_IN + idx];
        yf[idx] = v;
        yh[idx] = (_Float16)v;
    }

    for (int l = 0; l < NLAYER; ++l) {
        const _Float16* WL = Wpack + (size_t)l * PLAY;
#pragma unroll 1
        for (int net = 0; net < 2; ++net) {
            const _Float16* W0 = WL + net * PNET;
            const _Float16* W1 = W0 + PW0;
            const _Float16* W2 = W0 + PW0 + PW1;
            const float* b0 = (net ? sb0 : lb0) + l * NHID;
            const float* b1 = (net ? sb1 : lb1) + l * NHID;
            const float* b2 = (net ? sb2 : lb2) + l * D_IN;

            hidden_stage<D_IN, NHID>(W0, b0, yh, h0, lane);   // [32,64]x[64,256]
            hidden_stage<NHID, NHID>(W1, b1, h0, h1, lane);   // [32,256]x[256,256]

            // Final stage: [32,256] x [256,64], no relu.
#pragma unroll 1
            for (int nt = 0; nt < D_IN / 16; ++nt) {
                const int col = (lane & 15) + nt * 16;
                const float bb = b2[col];
                v8f acc0 = {bb, bb, bb, bb, bb, bb, bb, bb};
                v8f acc1 = acc0;
                const _Float16* Wb = W2 + ((size_t)nt * (NHID / 32)) * 512 + lane * 16;
#pragma unroll
                for (int ks = 0; ks < NHID / 32; ++ks) {
                    const v16h b  = *(const v16h*)(Wb + (size_t)ks * 512);
                    const v16h a0 = a_frag(h1, NHID, 0,  ks, lane);
                    const v16h a1 = a_frag(h1, NHID, 16, ks, lane);
                    acc0 = wmma16(a0, b, acc0);
                    acc1 = wmma16(a1, b, acc1);
                }
                const int rbase = (lane >> 4) * 8;
                if (net == 0) {
#pragma unroll
                    for (int v = 0; v < 8; ++v) {
                        locb[(size_t)(rbase + v) * D_IN + col]      = acc0[v];
                        locb[(size_t)(rbase + v + 16) * D_IN + col] = acc1[v];
                    }
                } else {
                    // Coupling: y = exp(-scale) * (y - loc)
#pragma unroll
                    for (int v = 0; v < 8; ++v) {
                        const int i0 = (rbase + v) * D_IN + col;
                        const int i1 = (rbase + v + 16) * D_IN + col;
                        const float n0 = __expf(-acc0[v]) * (yf[i0] - locb[i0]);
                        const float n1 = __expf(-acc1[v]) * (yf[i1] - locb[i1]);
                        yf[i0] = n0; yh[i0] = (_Float16)n0;
                        yf[i1] = n1; yh[i1] = (_Float16)n1;
                    }
                }
            }
        }
    }

    // Write back the wave's 32x64 output tile (f32).
#pragma unroll
    for (int i = 0; i < MROWS * D_IN / 32; ++i) {
        const int idx = i * 32 + lane;
        out[(size_t)row0 * D_IN + idx] = yf[idx];
    }
}

// Fold mask into weight, convert to f16, and swizzle into WMMA B-fragment
// order: dst[((ntile*(K/32) + kstep)*32 + lane)*16 + h], where
// col = (lane&15)+16*ntile, k = 32*kstep + 16*(lane>=16) + h.
__global__ void mask_pack(const float* __restrict__ W, const float* __restrict__ M,
                          _Float16* __restrict__ dst, int cnt, int K,
                          int dstOff, int total) {
    const int i = blockIdx.x * 256 + threadIdx.x;
    if (i >= total) return;
    const int l = i / cnt;
    const int j = i - l * cnt;
    const int n = j / K;
    const int k = j - n * K;
    const int ntile = n >> 4;
    const int colin = n & 15;
    const int kstep = k >> 5;
    const int kr    = k & 31;
    const int lane  = colin + ((kr >> 4) << 4);
    const int h     = kr & 15;
    const int dstj  = ((ntile * (K >> 5) + kstep) * 32 + lane) * 16 + h;
    dst[(size_t)l * PLAY + dstOff + dstj] = (_Float16)(W[i] * M[i]);
}

extern "C" void kernel_launch(void* const* d_in, const int* in_sizes, int n_in,
                              void* d_out, int out_size, void* d_ws, size_t ws_size,
                              hipStream_t stream) {
    const float* u   = (const float*)d_in[0];
    const float* lW0 = (const float*)d_in[1];
    const float* lb0 = (const float*)d_in[2];
    const float* lW1 = (const float*)d_in[3];
    const float* lb1 = (const float*)d_in[4];
    const float* lW2 = (const float*)d_in[5];
    const float* lb2 = (const float*)d_in[6];
    const float* sW0 = (const float*)d_in[7];
    const float* sb0 = (const float*)d_in[8];
    const float* sW1 = (const float*)d_in[9];
    const float* sb1 = (const float*)d_in[10];
    const float* sW2 = (const float*)d_in[11];
    const float* sb2 = (const float*)d_in[12];
    const float* M0  = (const float*)d_in[13];
    const float* M1  = (const float*)d_in[14];
    const float* M2  = (const float*)d_in[15];
    float* out = (float*)d_out;

    _Float16* Wpack = (_Float16*)d_ws;   // NLAYER * PLAY halves = ~6.3 MB

    struct Job { const float* W; const float* M; int cnt; int K; int off; };
    const Job jobs[6] = {
        {lW0, M0, PW0, D_IN, 0},
        {lW1, M1, PW1, NHID, PW0},
        {lW2, M2, PW2, NHID, PW0 + PW1},
        {sW0, M0, PW0, D_IN, PNET},
        {sW1, M1, PW1, NHID, PNET + PW0},
        {sW2, M2, PW2, NHID, PNET + PW0 + PW1},
    };
    for (int j = 0; j < 6; ++j) {
        const int total = NLAYER * jobs[j].cnt;
        mask_pack<<<(total + 255) / 256, 256, 0, stream>>>(
            jobs[j].W, jobs[j].M, Wpack, jobs[j].cnt, jobs[j].K, jobs[j].off, total);
    }

    const int rows = in_sizes[0] / D_IN;   // 32768
    const int blocks = rows / MROWS;       // 1024
    flow_kernel<<<blocks, 32, WAVE_LDS, stream>>>(
        u, lb0, lb1, lb2, sb0, sb1, sb2, Wpack, out);
}
